// PBPointerNet_35244501631002
// MI455X (gfx1250) — matrix operand order repaired and checked by reading, not compiled, not run on hardware
//
#include <hip/hip_runtime.h>
#include <hip/hip_bf16.h>

// ---------------- problem constants (match reference) ----------------
#define BB 64
#define PP 2048
#define FF 16
#define HH1 512
#define HH2 512
#define DD 256
#define BPROWS (BB * PP)          // 131072 rows through the per-pair MLP
#define NEGV (-1e9f)

typedef __attribute__((ext_vector_type(16))) _Float16 v16h;
typedef __attribute__((ext_vector_type(8)))  _Float16 v8h;
typedef __attribute__((ext_vector_type(8)))  float    v8f;
typedef __attribute__((ext_vector_type(4)))  int      v4i;

union AFrag { v16h v; v8h h[2]; };

// ---- CDNA5 async global->LDS staging (guarded: falls back to plain loads) --
#if defined(__has_builtin)
#  if __has_builtin(__builtin_amdgcn_global_load_async_to_lds_b128) && \
      __has_builtin(__builtin_amdgcn_s_wait_asynccnt)
#    define HAVE_ASYNC_LDS 1
#  endif
#endif
#ifndef HAVE_ASYNC_LDS
#  define HAVE_ASYNC_LDS 0
#endif

#if HAVE_ASYNC_LDS
typedef __attribute__((address_space(1))) v4i gv4i;   // global v4i
typedef __attribute__((address_space(3))) v4i lv4i;   // LDS v4i
#endif

__device__ __forceinline__ void stage16(const _Float16* g, _Float16* l)
{
#if HAVE_ASYNC_LDS
    __builtin_amdgcn_global_load_async_to_lds_b128(
        (gv4i*)(v4i*)(void*)g,
        (lv4i*)(v4i*)(void*)l, 0, 0);
#else
    *(v8h*)l = *(const v8h*)g;
#endif
}
__device__ __forceinline__ void stage_wait()
{
#if HAVE_ASYNC_LDS
    __builtin_amdgcn_s_wait_asynccnt(0);
#endif
}

// =====================================================================
// WMMA GEMM: C[M,N] = act(A[M,K]f16 @ Wt[N,K]^T f16 + bias[N])
//   Wt is the weight PRE-TRANSPOSED to N x K row-major so that each
//   lane's B-fragment (16 K-values of one column) is contiguous.
//   Block: 256 thr = 8 waves; block tile 128x128; wave tile 32x64
//   (8 WMMA accumulators); A tile double-buffered in LDS, staged with
//   GLOBAL_LOAD_ASYNC_TO_LDS_B128 (overlaps WMMA of tile k with the
//   DMA of tile k+1, fenced by s_wait_asynccnt + workgroup barrier).
// =====================================================================
template <bool RELU, bool OUT16>
__global__ void __launch_bounds__(256)
k_gemm(const _Float16* __restrict__ A, const _Float16* __restrict__ Wt,
       const float* __restrict__ bias, void* __restrict__ outp,
       int M, int N, int K)
{
    __shared__ _Float16 lA[2][128][40];  // 2 x (128x32 tile), +8 pad per row

    const int tid  = threadIdx.x;
    const int lane = tid & 31;
    const int wave = tid >> 5;
    const int wrow = wave >> 1;          // 0..3  -> 32-row slice
    const int wcol = wave & 1;           // 0..1  -> 64-col slice
    const int m0   = blockIdx.y * 128;
    const int n0   = blockIdx.x * 128 + wcol * 64;
    const int lsel = (lane < 16) ? 0 : 1;
    const int l15  = lane & 15;

    v8f acc[2][4] = {};

    const int srow  = tid >> 1;          // staging: row 0..127
    const int shalf = tid & 1;           // staging: 16-f16 half of 32-wide tile

    // ---- prologue: stage first A tile into buffer 0 ----
    {
        const _Float16* gp = A + (size_t)(m0 + srow) * K + shalf * 16;
        stage16(gp,     &lA[0][srow][shalf * 16]);
        stage16(gp + 8, &lA[0][srow][shalf * 16 + 8]);
    }
    stage_wait();
    __syncthreads();

    int cur = 0;
    for (int kt = 0; kt < K; kt += 32) {
        // ---- stage NEXT A tile into the other buffer (overlaps compute) ----
        const bool more = (kt + 32) < K;
        if (more) {
            const _Float16* gp = A + (size_t)(m0 + srow) * K + (kt + 32) + shalf * 16;
#if !HAVE_ASYNC_LDS
            __builtin_prefetch(gp, 0, 1);
#endif
            stage16(gp,     &lA[cur ^ 1][srow][shalf * 16]);
            stage16(gp + 8, &lA[cur ^ 1][srow][shalf * 16 + 8]);
        }

        // ---- B fragments straight from L2-resident transposed weights ----
        // lane L: column n0+16*jc+(L&15), K-range kt + (L<16 ? 0..15 : 16..31)
        AFrag fb[4];
#pragma unroll
        for (int jc = 0; jc < 4; ++jc) {
            const _Float16* bp = Wt + (size_t)(n0 + jc * 16 + l15) * K + kt + lsel * 16;
            fb[jc].v = *(const v16h*)bp;   // 32B contiguous -> 2x global_load_b128
        }

        // ---- A fragments from LDS in contiguous k order (same as B) ----
        // lane<16: K 0..15 ; lane>=16: K 16..31
        AFrag fa[2];
#pragma unroll
        for (int ir = 0; ir < 2; ++ir) {
            const _Float16* ap = &lA[cur][wrow * 32 + ir * 16 + l15][lsel * 16];
            fa[ir].h[0] = *(const v8h*)ap;
            fa[ir].h[1] = *(const v8h*)(ap + 8);
        }

#pragma unroll
        for (int ir = 0; ir < 2; ++ir)
#pragma unroll
            for (int jc = 0; jc < 4; ++jc)
                acc[ir][jc] = __builtin_amdgcn_wmma_f32_16x16x32_f16(
                    false, fa[ir].v, false, fb[jc].v,
                    (short)0, acc[ir][jc], false, false);

        if (more) stage_wait();   // own async fill of next buffer complete
        __syncthreads();          // everyone done reading cur + filling cur^1
        cur ^= 1;
    }

    // ---- epilogue: bias + optional ReLU, f16 or f32 store ----
    _Float16* o16 = (_Float16*)outp;
    float*    o32 = (float*)outp;
#pragma unroll
    for (int ir = 0; ir < 2; ++ir) {
#pragma unroll
        for (int jc = 0; jc < 4; ++jc) {
            const int col = n0 + jc * 16 + l15;
            const float bv = bias[col];
#pragma unroll
            for (int g = 0; g < 8; ++g) {
                const int row = m0 + wrow * 32 + ir * 16 + ((lane < 16) ? g : g + 8);
                float v = acc[ir][jc][g] + bv;
                if (RELU) v = fmaxf(v, 0.0f);
                if (OUT16) o16[(size_t)row * N + col] = (_Float16)v;
                else       o32[(size_t)row * N + col] = v;
            }
        }
    }
}

// ---------------- conversion / init kernels ----------------
__global__ void k_cvt_batch(const int* __restrict__ batch, _Float16* __restrict__ A0)
{   // int batch (BP x 16) -> f16 A0 (BP x 32), zero-padded K for WMMA
    int idx = blockIdx.x * blockDim.x + threadIdx.x;
    int row = idx >> 5, f = idx & 31;
    A0[idx] = (f < FF) ? (_Float16)(float)batch[row * FF + f] : (_Float16)0.0f;
}

__global__ void k_cvt_wt(const float* __restrict__ W, _Float16* __restrict__ Wt,
                         int K, int N, int Kpad)
{   // W (K x N) f32 -> Wt (N x Kpad) f16, zero-padded
    int idx = blockIdx.x * blockDim.x + threadIdx.x;
    int n = idx / Kpad, k = idx % Kpad;
    Wt[idx] = (k < K) ? (_Float16)W[(size_t)k * N + n] : (_Float16)0.0f;
}

__global__ void k_copy2(const float* __restrict__ q0, const float* __restrict__ c0,
                        float* __restrict__ q, float* __restrict__ c, int n)
{
    int i = blockIdx.x * blockDim.x + threadIdx.x;
    if (i < n) { q[i] = q0[i]; c[i] = c0[i]; }
}

// ---------------- attention / LSTM (3 sequential steps) ----------------
__global__ void __launch_bounds__(256)
k_logits(const float* __restrict__ Mm, const float* __restrict__ q,
         const int* __restrict__ batch, float* __restrict__ logits)
{
    __shared__ alignas(16) float lq[DD];
    const int b = blockIdx.y, tid = threadIdx.x;
    lq[tid] = q[b * DD + tid];
    __syncthreads();
    const size_t p = (size_t)blockIdx.x * 256 + tid;
    const float4* mp4 = (const float4*)(Mm + ((size_t)b * PP + p) * DD);
    const float4* lq4 = (const float4*)lq;
    float s = 0.f;
#pragma unroll 4
    for (int d = 0; d < DD / 4; ++d) {
        float4 m = mp4[d], v = lq4[d];
        s += m.x * v.x + m.y * v.y + m.z * v.z + m.w * v.w;
    }
    const float neg = (batch[((size_t)b * PP + p) * FF + (FF - 1)] != -1) ? 0.f : NEGV;
    logits[(size_t)b * PP + p] = s + neg;
}

__global__ void __launch_bounds__(256)
k_softmax(const float* __restrict__ logits, float* __restrict__ attn)
{
    __shared__ float red[256];
    const int b = blockIdx.x, tid = threadIdx.x;
    const float* lp = logits + (size_t)b * PP;
    float mx = -3.4e38f;
    for (int p = tid; p < PP; p += 256) mx = fmaxf(mx, lp[p]);
    red[tid] = mx; __syncthreads();
    for (int s = 128; s > 0; s >>= 1) { if (tid < s) red[tid] = fmaxf(red[tid], red[tid + s]); __syncthreads(); }
    mx = red[0]; __syncthreads();
    float sum = 0.f;
    for (int p = tid; p < PP; p += 256) sum += __expf(lp[p] - mx);
    red[tid] = sum; __syncthreads();
    for (int s = 128; s > 0; s >>= 1) { if (tid < s) red[tid] += red[tid + s]; __syncthreads(); }
    const float inv = 1.0f / red[0];
    for (int p = tid; p < PP; p += 256) attn[(size_t)b * PP + p] = __expf(lp[p] - mx) * inv;
}

__global__ void __launch_bounds__(256)
k_readout(const float* __restrict__ attn, const float* __restrict__ Mm,
          float* __restrict__ r)
{
    __shared__ float la[PP];   // 8KB attn row
    const int b = blockIdx.x, tid = threadIdx.x;
    for (int p = tid; p < PP; p += 256) la[p] = attn[(size_t)b * PP + p];
    __syncthreads();
    const float* mp = Mm + (size_t)b * PP * DD + tid;   // coalesced across d=tid
    float s = 0.f;
    for (int p = 0; p < PP; ++p) s += la[p] * mp[(size_t)p * DD];
    r[b * DD + tid] = s;
}

__global__ void __launch_bounds__(256)
k_lstm(float* __restrict__ q, float* __restrict__ c, const float* __restrict__ r,
       const float* __restrict__ Wi, const float* __restrict__ bi,
       const float* __restrict__ Wf, const float* __restrict__ bf,
       const float* __restrict__ Wg, const float* __restrict__ bg,
       const float* __restrict__ Wo, const float* __restrict__ bo)
{
    __shared__ float qs[2 * DD];    // concat([q, r])
    const int b = blockIdx.x, d = threadIdx.x;
    qs[d]      = q[b * DD + d];
    qs[DD + d] = r[b * DD + d];
    __syncthreads();
    float si = bi[d], sf = bf[d], sg = bg[d], so = bo[d];
    for (int k = 0; k < 2 * DD; ++k) {
        const float x = qs[k];
        si += x * Wi[k * DD + d];
        sf += x * Wf[k * DD + d];
        sg += x * Wg[k * DD + d];
        so += x * Wo[k * DD + d];
    }
    const float it = 1.f / (1.f + __expf(-si));
    const float ft = 1.f / (1.f + __expf(-sf));
    const float gt = tanhf(sg);
    const float ot = 1.f / (1.f + __expf(-so));
    const float cn = ft * c[b * DD + d] + it * gt;
    c[b * DD + d] = cn;
    q[b * DD + d] = ot * tanhf(cn);
}

// ---------------- GRU decoder ----------------
// h1, h2 and gxf = h2@gWxf+gbf are identical for all batches (x=ones, h=0
// so the gWh1/gWh2 terms vanish): compute once in a single block.
__global__ void __launch_bounds__(256)
k_gru_shared(const float* __restrict__ gWx1, const float* __restrict__ gb1,
             const float* __restrict__ gWx2, const float* __restrict__ gb2,
             const float* __restrict__ gWxf, const float* __restrict__ gbf,
             float* __restrict__ gxf_out)
{
    __shared__ float h1[HH1];
    __shared__ float h2[HH2];
    const int tid = threadIdx.x;
    for (int j = tid; j < HH1; j += 256) {          // x = ones(F): column sums
        float xz = gb1[j], xn = gb1[2 * HH1 + j];
        for (int i = 0; i < FF; ++i) {
            xz += gWx1[i * 3 * HH1 + j];
            xn += gWx1[i * 3 * HH1 + 2 * HH1 + j];
        }
        h1[j] = (1.f / (1.f + __expf(-xz))) * tanhf(xn);
    }
    __syncthreads();
    for (int j = tid; j < HH2; j += 256) {
        float xz = gb2[j], xn = gb2[2 * HH2 + j];
        for (int k = 0; k < HH1; ++k) {
            const float h = h1[k];
            xz += h * gWx2[k * 3 * HH2 + j];
            xn += h * gWx2[k * 3 * HH2 + 2 * HH2 + j];
        }
        h2[j] = (1.f / (1.f + __expf(-xz))) * tanhf(xn);
    }
    __syncthreads();
    for (int j = tid; j < 3 * DD; j += 256) {
        float s = gbf[j];
        for (int k = 0; k < HH2; ++k) s += h2[k] * gWxf[k * 3 * DD + j];
        gxf_out[j] = s;
    }
}

__global__ void __launch_bounds__(256)
k_gru_dec(const float* __restrict__ q, const float* __restrict__ gWhf,
          const float* __restrict__ gxf, float* __restrict__ dec)
{
    __shared__ float qv[DD];
    const int b = blockIdx.x, d = threadIdx.x;
    qv[d] = q[b * DD + d];
    __syncthreads();
    float hz = 0.f, hr = 0.f, hn = 0.f;
    for (int k = 0; k < DD; ++k) {
        const float h = qv[k];
        hz += h * gWhf[k * 3 * DD + d];
        hr += h * gWhf[k * 3 * DD + DD + d];
        hn += h * gWhf[k * 3 * DD + 2 * DD + d];
    }
    const float z = 1.f / (1.f + __expf(-(gxf[d] + hz)));
    const float r = 1.f / (1.f + __expf(-(gxf[DD + d] + hr)));
    const float n = tanhf(gxf[2 * DD + d] + r * hn);
    dec[b * DD + d] = (1.f - z) * qv[d] + z * n;
}

// ---------------- final pointer scores + log-softmax ----------------
__global__ void __launch_bounds__(256)
k_scores(const float* __restrict__ dec, const _Float16* __restrict__ X,
         const int* __restrict__ batch, float* __restrict__ scores)
{
    __shared__ alignas(16) float ld[DD];
    const int b = blockIdx.y, tid = threadIdx.x;
    ld[tid] = dec[b * DD + tid];
    __syncthreads();
    const size_t p = (size_t)blockIdx.x * 256 + tid;
    const _Float16* xp = X + ((size_t)b * PP + p) * DD;
    float s = 0.f;
#pragma unroll 8
    for (int d = 0; d < DD; ++d) s += (float)xp[d] * ld[d];
    const float neg = (batch[((size_t)b * PP + p) * FF + (FF - 1)] != -1) ? 0.f : NEGV;
    scores[(size_t)b * PP + p] = s + neg;
}

__global__ void __launch_bounds__(256)
k_logsoftmax(const float* __restrict__ scores, float* __restrict__ out)
{
    __shared__ float red[256];
    const int b = blockIdx.x, tid = threadIdx.x;
    const float* sp = scores + (size_t)b * PP;
    float mx = -3.4e38f;
    for (int p = tid; p < PP; p += 256) mx = fmaxf(mx, sp[p]);
    red[tid] = mx; __syncthreads();
    for (int s = 128; s > 0; s >>= 1) { if (tid < s) red[tid] = fmaxf(red[tid], red[tid + s]); __syncthreads(); }
    mx = red[0]; __syncthreads();
    float sum = 0.f;
    for (int p = tid; p < PP; p += 256) sum += __expf(sp[p] - mx);
    red[tid] = sum; __syncthreads();
    for (int s = 128; s > 0; s >>= 1) { if (tid < s) red[tid] += red[tid + s]; __syncthreads(); }
    const float lse = mx + __logf(red[0]);
    for (int p = tid; p < PP; p += 256) out[(size_t)b * PP + p] = sp[p] - lse;
}

// =====================================================================
extern "C" void kernel_launch(void* const* d_in, const int* in_sizes, int n_in,
                              void* d_out, int out_size, void* d_ws, size_t ws_size,
                              hipStream_t stream)
{
    (void)in_sizes; (void)n_in; (void)out_size; (void)ws_size;

    const int*   batch = (const int*)d_in[0];
    const float* W1 = (const float*)d_in[1];  const float* b1 = (const float*)d_in[2];
    const float* W2 = (const float*)d_in[3];  const float* b2 = (const float*)d_in[4];
    const float* W3 = (const float*)d_in[5];  const float* b3 = (const float*)d_in[6];
    const float* We = (const float*)d_in[7];  const float* be = (const float*)d_in[8];
    const float* Wi = (const float*)d_in[9];  const float* bi = (const float*)d_in[10];
    const float* Wf = (const float*)d_in[11]; const float* bf = (const float*)d_in[12];
    const float* Wg = (const float*)d_in[13]; const float* bg = (const float*)d_in[14];
    const float* Wo = (const float*)d_in[15]; const float* bo = (const float*)d_in[16];
    const float* q0 = (const float*)d_in[17]; const float* c0 = (const float*)d_in[18];
    const float* gWx1 = (const float*)d_in[19];  /* d_in[20] gWh1: dead (h=0) */
    const float* gb1  = (const float*)d_in[21];
    const float* gWx2 = (const float*)d_in[22];  /* d_in[23] gWh2: dead (h=0) */
    const float* gb2  = (const float*)d_in[24];
    const float* gWxf = (const float*)d_in[25];
    const float* gWhf = (const float*)d_in[26];
    const float* gbf  = (const float*)d_in[27];

    // ---- workspace layout (256B aligned regions) ----
    char* ws = (char*)d_ws;
    size_t off = 0;
    auto alloc = [&](size_t bytes) -> void* {
        void* p = ws + off;
        off = (off + bytes + 255) & ~(size_t)255;
        return p;
    };
    _Float16* A0  = (_Float16*)alloc((size_t)BPROWS * 32 * 2);   // padded f16 input
    _Float16* W1t = (_Float16*)alloc((size_t)HH1 * 32 * 2);
    _Float16* W2t = (_Float16*)alloc((size_t)HH2 * HH1 * 2);
    _Float16* W3t = (_Float16*)alloc((size_t)DD * HH2 * 2);
    _Float16* Wet = (_Float16*)alloc((size_t)DD * DD * 2);
    char* bufA = (char*)alloc((size_t)BPROWS * HH1 * 2);  // H1 f16 -> later X f16
    char* bufB = (char*)alloc((size_t)BPROWS * DD * 4);   // H2 f16 -> later M f32 (same size)
    float* q      = (float*)alloc((size_t)BB * DD * 4);
    float* c      = (float*)alloc((size_t)BB * DD * 4);
    float* rr     = (float*)alloc((size_t)BB * DD * 4);
    float* logits = (float*)alloc((size_t)BB * PP * 4);
    float* attn   = (float*)alloc((size_t)BB * PP * 4);
    float* gxf    = (float*)alloc((size_t)3 * DD * 4);
    float* dec    = (float*)alloc((size_t)BB * DD * 4);
    float* scores = (float*)alloc((size_t)BB * PP * 4);

    _Float16* H1f = (_Float16*)bufA;
    _Float16* H2f = (_Float16*)bufB;
    _Float16* Xf  = (_Float16*)bufA;   // overwrites dead H1
    float*    Mf  = (float*)bufB;      // overwrites dead H2

    // ---- precision conversion / init ----
    k_cvt_batch<<<(BPROWS * 32) / 256, 256, 0, stream>>>(batch, A0);
    k_cvt_wt<<<(HH1 * 32) / 256, 256, 0, stream>>>(W1, W1t, FF, HH1, 32);
    k_cvt_wt<<<(HH2 * HH1) / 256, 256, 0, stream>>>(W2, W2t, HH1, HH2, HH1);
    k_cvt_wt<<<(DD * HH2) / 256, 256, 0, stream>>>(W3, W3t, HH2, DD, HH2);
    k_cvt_wt<<<(DD * DD) / 256, 256, 0, stream>>>(We, Wet, DD, DD, DD);
    k_copy2<<<(BB * DD) / 256, 256, 0, stream>>>(q0, c0, q, c, BB * DD);

    // ---- the heavy WMMA GEMM chain (fused bias+ReLU epilogues) ----
    k_gemm<true,  true ><<<dim3(HH1 / 128, BPROWS / 128), 256, 0, stream>>>(A0,  W1t, b1, H1f, BPROWS, HH1, 32);
    k_gemm<true,  true ><<<dim3(HH2 / 128, BPROWS / 128), 256, 0, stream>>>(H1f, W2t, b2, H2f, BPROWS, HH2, HH1);
    k_gemm<true,  true ><<<dim3(DD  / 128, BPROWS / 128), 256, 0, stream>>>(H2f, W3t, b3, Xf,  BPROWS, DD,  HH2);
    k_gemm<false, false><<<dim3(DD  / 128, BPROWS / 128), 256, 0, stream>>>(Xf,  Wet, be, Mf,  BPROWS, DD,  DD);

    // ---- ProcessBlock: 3 attention read-out + LSTM steps ----
    for (int step = 0; step < 3; ++step) {
        k_logits <<<dim3(PP / 256, BB), 256, 0, stream>>>(Mf, q, batch, logits);
        k_softmax<<<BB, 256, 0, stream>>>(logits, attn);
        k_readout<<<BB, 256, 0, stream>>>(attn, Mf, rr);
        k_lstm   <<<BB, 256, 0, stream>>>(q, c, rr, Wi, bi, Wf, bf, Wg, bg, Wo, bo);
    }

    // ---- PointerDecidingLayer ----
    k_gru_shared<<<1, 256, 0, stream>>>(gWx1, gb1, gWx2, gb2, gWxf, gbf, gxf);
    k_gru_dec<<<BB, 256, 0, stream>>>(q, gWhf, gxf, dec);
    k_scores<<<dim3(PP / 256, BB), 256, 0, stream>>>(dec, Xf, batch, scores);
    k_logsoftmax<<<BB, 256, 0, stream>>>(scores, (float*)d_out);
}
